// CTRNN_730144440950
// MI455X (gfx1250) — matrix operand order repaired and checked by reading, not compiled, not run on hardware
//
#include <hip/hip_runtime.h>

// ---- problem dims (fixed by reference) ----
#define B_TOT   256
#define T_STEPS 2000
#define H       256
#define IN      64
#define NT      8
#define DIN     72      // IN + NT
#define OUTN    32
#define MTILE   16      // batch rows per workgroup
#define ONEMA   0.9f    // 1 - alpha
#define ALPHAF  0.1f

// ---- LDS row pitches (halves). 264*2B = 528B = 132 words ≡ 4 (mod 64 banks)
#define WOP  264   // W_out rows [OUTN][WOP]
#define AP   264   // relu(h)/h bf16 tiles [MTILE][AP]
#define XP   72    // x_t bf16 tile [MTILE][XP]

typedef __bf16 bf16_t;
typedef bf16_t v16bf __attribute__((ext_vector_type(16)));
typedef bf16_t v8bf  __attribute__((ext_vector_type(8)));
typedef float  v8f   __attribute__((ext_vector_type(8)));

static __device__ __forceinline__ v16bf cat16(v8bf lo, v8bf hi) {
  return __builtin_shufflevector(lo, hi, 0,1,2,3,4,5,6,7,8,9,10,11,12,13,14,15);
}

// A fragment, 16-bit A 16x32: lanes 0-15 m=lane, K=k0+0..15;
// lanes 16-31 m=lane-16, K=k0+16..31 (contiguous, matches B fragments).
static __device__ __forceinline__ v16bf ld_afrag(const bf16_t* rowbase, int k0, int lhalf) {
  const bf16_t* p = rowbase + k0 + lhalf * 16;
  v8bf lo = *(const v8bf*)(p);
  v8bf hi = *(const v8bf*)(p + 8);
  return cat16(lo, hi);
}

// B fragment from LDS, 16-bit B 32x16: lanes 0-15 hold K=k0..k0+15 of col n0+lane,
// lanes 16-31 hold K=k0+16..k0+31 (contiguous 16 halves per lane).
static __device__ __forceinline__ v16bf ld_bfrag_lds(const bf16_t* w, int pitch, int n0,
                                                     int k0, int lmod, int lhalf) {
  const bf16_t* p = w + (n0 + lmod) * pitch + k0 + lhalf * 16;
  v8bf lo = *(const v8bf*)(p);
  v8bf hi = *(const v8bf*)(p + 8);
  return cat16(lo, hi);
}

// B fragment straight from global fp32 weights (one-time, converted to bf16).
static __device__ __forceinline__ v16bf ld_bfrag_global(const float* w, int ldw, int n0,
                                                        int k0, int lmod, int lhalf) {
  const float* p = w + (size_t)(n0 + lmod) * ldw + k0 + 16 * lhalf;
  v16bf r;
#pragma unroll
  for (int i = 0; i < 16; ++i) r[i] = (bf16_t)p[i];
  return r;
}

static __device__ __forceinline__ v8f wmma_bf16(v16bf a, v16bf b, v8f c) {
  return __builtin_amdgcn_wmma_f32_16x16x32_bf16(false, a, false, b, (short)0, c,
                                                 false, false);
}

__global__ __launch_bounds__(256, 1)
void ctrnn_scan_kernel(const float* __restrict__ x,
                       const float* __restrict__ task_id,
                       const float* __restrict__ W_in,
                       const float* __restrict__ b_in,
                       const float* __restrict__ W_hh,
                       const float* __restrict__ b_hh,
                       const float* __restrict__ W_out,
                       const float* __restrict__ b_out,
                       float* __restrict__ out) {
  extern __shared__ bf16_t smem[];
  bf16_t* s_wout = smem;                     // [OUTN][WOP]
  bf16_t* s_a    = s_wout + OUTN * WOP;      // [2][MTILE][AP]  relu(h) bf16
  bf16_t* s_hb   = s_a    + 2 * MTILE * AP;  // [2][MTILE][AP]  h bf16
  bf16_t* s_xa   = s_hb   + 2 * MTILE * AP;  // [2][MTILE][XP]  x_t bf16

  const int tid   = threadIdx.x;
  const int lane  = tid & 31;
  const int wave  = tid >> 5;
  const int lmod  = lane & 15;
  const int lhalf = lane >> 4;
  const int b0    = blockIdx.x * MTILE;
  const int n0w   = wave * 32;               // this wave's 32 hidden columns

  // ---- one-time staging ----
  for (int i = tid; i < OUTN * H; i += 256)
    s_wout[(i >> 8) * WOP + (i & 255)] = (bf16_t)W_out[i];
  for (int i = tid; i < MTILE * AP; i += 256)   // "prev" buffer (idx 1): relu(h0)==0
    s_a[MTILE * AP + i] = (bf16_t)0.f;

  // Loop-invariant B fragments live in registers for the whole scan.
  v16bf whf[8][2];   // W_hh  [k0/32][n-tile]
  v16bf wif[2][2];   // W_in64[k0/32][n-tile]
#pragma unroll
  for (int kk = 0; kk < 8; ++kk) {
    whf[kk][0] = ld_bfrag_global(W_hh, H, n0w,      kk * 32, lmod, lhalf);
    whf[kk][1] = ld_bfrag_global(W_hh, H, n0w + 16, kk * 32, lmod, lhalf);
  }
#pragma unroll
  for (int kk = 0; kk < 2; ++kk) {
    wif[kk][0] = ld_bfrag_global(W_in, DIN, n0w,      kk * 32, lmod, lhalf);
    wif[kk][1] = ld_bfrag_global(W_in, DIN, n0w + 16, kk * 32, lmod, lhalf);
  }

  // ---- register-resident state: h and C-init (task projection + biases) ----
  v8f hreg[2], cinit[2];
#pragma unroll
  for (int tt = 0; tt < 2; ++tt) {
    const int n0 = n0w + tt * 16;
#pragma unroll
    for (int r = 0; r < 8; ++r) {
      const int m = r + 8 * lhalf;
      const int n = n0 + lmod;
      float c = b_in[n] + b_hh[n];
#pragma unroll
      for (int j = 0; j < NT; ++j)
        c += task_id[(b0 + m) * NT + j] * W_in[n * DIN + IN + j];
      cinit[tt][r] = c;
      hreg[tt][r]  = 0.f;
    }
  }
  const float boutv = b_out[(wave < 2 ? wave * 16 : 0) + lmod];

  for (int t = 0; t < T_STEPS; ++t) {
    const int cur  = t & 1;
    const int prev = cur ^ 1;

    // ---- stage x_t tile as bf16 into buffer `cur` (4 elems/thread) ----
    {
      const int m = tid >> 4;
      const int d = (tid & 15) * 4;
      const float4 v =
          *(const float4*)(x + ((size_t)(b0 + m) * T_STEPS + t) * IN + d);
      bf16_t* dst = s_xa + cur * MTILE * XP + m * XP + d;
      dst[0] = (bf16_t)v.x; dst[1] = (bf16_t)v.y;
      dst[2] = (bf16_t)v.z; dst[3] = (bf16_t)v.w;
      if (t + 1 < T_STEPS)
        __builtin_prefetch(x + ((size_t)(b0 + m) * T_STEPS + (t + 1)) * IN + d, 0, 0);
    }
    __syncthreads();  // single barrier per step: all writes(t-1) + xa(t) visible

    // ---- recurrence: rec = c_init + x_t @ Win64^T + relu(h_{t-1}) @ Whh^T ----
    v8f acc0 = cinit[0];
    v8f acc1 = cinit[1];
    const bf16_t* xrow = s_xa + cur * MTILE * XP + lmod * XP;
#pragma unroll
    for (int kk = 0; kk < 2; ++kk) {
      v16bf af = ld_afrag(xrow, kk * 32, lhalf);
      acc0 = wmma_bf16(af, wif[kk][0], acc0);
      acc1 = wmma_bf16(af, wif[kk][1], acc1);
    }
    const bf16_t* arow = s_a + prev * MTILE * AP + lmod * AP;
#pragma unroll
    for (int kk = 0; kk < 8; ++kk) {
      v16bf af = ld_afrag(arow, kk * 32, lhalf);
      acc0 = wmma_bf16(af, whf[kk][0], acc0);
      acc1 = wmma_bf16(af, whf[kk][1], acc1);
    }
#pragma unroll
    for (int r = 0; r < 8; ++r) {
      hreg[0][r] = ONEMA * hreg[0][r] + ALPHAF * acc0[r];
      hreg[1][r] = ONEMA * hreg[1][r] + ALPHAF * acc1[r];
    }

    // ---- publish bf16 h / relu(h) into buffer `cur` ----
#pragma unroll
    for (int tt = 0; tt < 2; ++tt) {
      const int n0 = n0w + tt * 16;
#pragma unroll
      for (int r = 0; r < 8; ++r) {
        const int m = r + 8 * lhalf;
        const float hv = (tt == 0) ? hreg[0][r] : hreg[1][r];
        s_hb[cur * MTILE * AP + m * AP + n0 + lmod] = (bf16_t)hv;
        s_a [cur * MTILE * AP + m * AP + n0 + lmod] = (bf16_t)fmaxf(hv, 0.f);
      }
    }

    // ---- deferred output projection for step t-1 (reads buffer `prev`) ----
    if (wave < 2 && t > 0) {
      const int o0 = wave * 16;
      v8f oacc;
#pragma unroll
      for (int r = 0; r < 8; ++r) oacc[r] = boutv;
      const bf16_t* hrow = s_hb + prev * MTILE * AP + lmod * AP;
#pragma unroll
      for (int kk = 0; kk < 8; ++kk) {
        v16bf af = ld_afrag(hrow, kk * 32, lhalf);
        oacc = wmma_bf16(af, ld_bfrag_lds(s_wout, WOP, o0, kk * 32, lmod, lhalf), oacc);
      }
#pragma unroll
      for (int r = 0; r < 8; ++r) {
        const int m = r + 8 * lhalf;
        out[((size_t)(b0 + m) * T_STEPS + (t - 1)) * OUTN + o0 + lmod] = oacc[r];
      }
    }
  }

  // ---- epilogue: output projection for final step T-1 ----
  __syncthreads();
  if (wave < 2) {
    const int fin = (T_STEPS - 1) & 1;
    const int o0 = wave * 16;
    v8f oacc;
#pragma unroll
    for (int r = 0; r < 8; ++r) oacc[r] = boutv;
    const bf16_t* hrow = s_hb + fin * MTILE * AP + lmod * AP;
#pragma unroll
    for (int kk = 0; kk < 8; ++kk) {
      v16bf af = ld_afrag(hrow, kk * 32, lhalf);
      oacc = wmma_bf16(af, ld_bfrag_lds(s_wout, WOP, o0, kk * 32, lmod, lhalf), oacc);
    }
#pragma unroll
    for (int r = 0; r < 8; ++r) {
      const int m = r + 8 * lhalf;
      out[((size_t)(b0 + m) * T_STEPS + (T_STEPS - 1)) * OUTN + o0 + lmod] = oacc[r];
    }
  }
}

extern "C" void kernel_launch(void* const* d_in, const int* in_sizes, int n_in,
                              void* d_out, int out_size, void* d_ws, size_t ws_size,
                              hipStream_t stream) {
  (void)in_sizes; (void)n_in; (void)out_size; (void)d_ws; (void)ws_size;
  const float* x       = (const float*)d_in[0];
  const float* task_id = (const float*)d_in[1];
  const float* W_in    = (const float*)d_in[2];
  const float* b_in    = (const float*)d_in[3];
  const float* W_hh    = (const float*)d_in[4];
  const float* b_hh    = (const float*)d_in[5];
  const float* W_out   = (const float*)d_in[6];
  const float* b_out   = (const float*)d_in[7];
  float* out = (float*)d_out;

  constexpr size_t LDS_HALVES = (size_t)OUTN * WOP + 4u * MTILE * AP +
                                2u * (size_t)MTILE * XP;
  constexpr size_t LDS_BYTES = LDS_HALVES * sizeof(bf16_t);  // ~55 KB

  dim3 grid(B_TOT / MTILE);   // 16 independent batch tiles
  dim3 block(256);            // 8 wave32 waves
  ctrnn_scan_kernel<<<grid, block, LDS_BYTES, stream>>>(
      x, task_id, W_in, b_in, W_hh, b_hh, W_out, b_out, out);
}
